// NODE_64587718197729
// MI455X (gfx1250) — matrix operand order repaired and checked by reading, not compiled, not run on hardware
//
#include <hip/hip_runtime.h>
#include <hip/hip_bf16.h>

// ---------------------------------------------------------------------------
// Neural ODE (Euler, T=128 steps) for MI455X / gfx1250.
// GEMMs run as f16 WMMA (v_wmma_f32_16x16x32_f16) with f32 accumulation.
// Weights converted+transposed to f16 once per launch (L2-resident, 192MB L2).
// Wave tile 32x32 (2x2 WMMA tiles): 8x16B loads per 4 WMMAs (2 loads/WMMA).
// Final-GEMM epilogue emits both f32 state and the f16 A-matrix for the next
// step, so the sequential chain is 5 kernels/step with no conversion pass.
// ---------------------------------------------------------------------------

typedef __attribute__((ext_vector_type(16))) _Float16 v16h;
typedef __attribute__((ext_vector_type(8)))  _Float16 v8h;
typedef __attribute__((ext_vector_type(8)))  float    v8f;

#define B_DIM 256
#define T_DIM 128
#define X_DIM 256
#define H_DIM 1024
#define NLAYERS 3

union FragU { v16h v; v8h h[2]; };

// ---------------------------------------------------------------------------
// Weight convert: W (K x N, f32, row-major) -> WT (N x K, f16, row-major)
// so WMMA B-fragments become contiguous 16B loads along K.
// ---------------------------------------------------------------------------
__global__ void k_transpose_to_f16(const float* __restrict__ W,
                                   _Float16* __restrict__ WT,
                                   int K, int N) {
  int idx = blockIdx.x * blockDim.x + threadIdx.x;
  if (idx >= K * N) return;
  int n = idx / K;
  int k = idx - n * K;
  WT[idx] = (_Float16)W[(size_t)k * N + n];
}

// y = x ; out[:, 0, :] = x ; A16y = (f16) x
__global__ void k_init(const float* __restrict__ x,
                       float* __restrict__ y,
                       float* __restrict__ out,
                       _Float16* __restrict__ a16y) {
  int idx = blockIdx.x * blockDim.x + threadIdx.x;
  if (idx >= B_DIM * X_DIM) return;
  int b = idx >> 8;           // / X_DIM
  int c = idx & (X_DIM - 1);
  float v = x[idx];
  y[idx] = v;
  a16y[idx] = (_Float16)v;
  out[((size_t)b * T_DIM) * X_DIM + c] = v;
}

// ---------------------------------------------------------------------------
// WMMA GEMM + bias + tanh -> f16:
//   Out[m][n] = (f16) tanh( sum_k A[m][k] * WT[n][k] + bias[n] )
// Block = 128 threads (4 waves, 2x2). WG tile 64x64; wave tile 32x32
// (2x2 WMMA 16x16 tiles -> each A/B fragment feeds two v_wmma).
// A fragment (16x32 f16): lane lm=lane&15 -> M row; lane half lh selects the
//   8-element K interleave (K: lh*8+0..7 then 16+lh*8+0..7) -> two 16B loads.
// B fragment (32x16 f16): lane lm -> N col; same K interleave as A -> two 16B loads.
// C/D (16x16 f32): c[r] -> M = r + 8*lh, N = lm.
// No divergence: EXEC all-ones for WMMA as required.
// ---------------------------------------------------------------------------
#define WMMA_F16(A_, B_, C_) \
  __builtin_amdgcn_wmma_f32_16x16x32_f16(false, (A_), false, (B_), (short)0, (C_), false, false)

template <int K, int N>
__global__ __launch_bounds__(128)
void k_gemm_tanh(const _Float16* __restrict__ A,
                 const _Float16* __restrict__ WT,
                 const float* __restrict__ bias,
                 _Float16* __restrict__ Out) {
  const int lane = threadIdx.x & 31;
  const int w    = threadIdx.x >> 5;       // 0..3
  const int lm   = lane & 15;
  const int lh   = lane >> 4;
  const int m0   = blockIdx.y * 64 + (w & 1) * 32;
  const int n0   = blockIdx.x * 64 + (w >> 1) * 32;

  const _Float16* a0r = A  + (size_t)(m0 + lm) * K + lh * 8;
  const _Float16* a1r = a0r + (size_t)16 * K;
  const _Float16* b0r = WT + (size_t)(n0 + lm) * K + lh * 8;
  const _Float16* b1r = b0r + (size_t)16 * K;

  v8f acc00 = {}, acc01 = {}, acc10 = {}, acc11 = {};
#pragma unroll 2
  for (int k0 = 0; k0 < K; k0 += 32) {
    FragU a0, a1, b0, b1;
    a0.h[0] = *(const v8h*)(a0r + k0);
    a0.h[1] = *(const v8h*)(a0r + k0 + 16);
    a1.h[0] = *(const v8h*)(a1r + k0);
    a1.h[1] = *(const v8h*)(a1r + k0 + 16);
    b0.h[0] = *(const v8h*)(b0r + k0);
    b0.h[1] = *(const v8h*)(b0r + k0 + 16);
    b1.h[0] = *(const v8h*)(b1r + k0);
    b1.h[1] = *(const v8h*)(b1r + k0 + 16);
    acc00 = WMMA_F16(a0.v, b0.v, acc00);
    acc01 = WMMA_F16(a0.v, b1.v, acc01);
    acc10 = WMMA_F16(a1.v, b0.v, acc10);
    acc11 = WMMA_F16(a1.v, b1.v, acc11);
  }

  const float bi0 = bias[n0 + lm];
  const float bi1 = bias[n0 + 16 + lm];
#pragma unroll
  for (int r = 0; r < 8; ++r) {
    const int m = m0 + r + 8 * lh;
    Out[(size_t)m * N + n0 + lm]             = (_Float16)tanhf(acc00[r] + bi0);
    Out[(size_t)m * N + n0 + 16 + lm]        = (_Float16)tanhf(acc01[r] + bi1);
    Out[(size_t)(m + 16) * N + n0 + lm]      = (_Float16)tanhf(acc10[r] + bi0);
    Out[(size_t)(m + 16) * N + n0 + 16 + lm] = (_Float16)tanhf(acc11[r] + bi1);
  }
}

// ---------------------------------------------------------------------------
// Final GEMM + fused Euler update:
//   f = A @ WT_out + b_out ; y_new = y + dt * f
//   out[:, t, :] = y_new ; A16y = (f16) y_new   (next step's A matrix)
// M = B_DIM, K = H_DIM, N = X_DIM. dt read from time_steps on device.
// ---------------------------------------------------------------------------
__global__ __launch_bounds__(128)
void k_gemm_final(const _Float16* __restrict__ A,
                  const _Float16* __restrict__ WT,
                  const float* __restrict__ bias,
                  const float* __restrict__ ts,
                  float* __restrict__ y,
                  float* __restrict__ out,
                  _Float16* __restrict__ a16y,
                  int t) {
  constexpr int K = H_DIM;
  constexpr int N = X_DIM;
  const int lane = threadIdx.x & 31;
  const int w    = threadIdx.x >> 5;
  const int lm   = lane & 15;
  const int lh   = lane >> 4;
  const int m0   = blockIdx.y * 64 + (w & 1) * 32;
  const int n0   = blockIdx.x * 64 + (w >> 1) * 32;

  const _Float16* a0r = A  + (size_t)(m0 + lm) * K + lh * 8;
  const _Float16* a1r = a0r + (size_t)16 * K;
  const _Float16* b0r = WT + (size_t)(n0 + lm) * K + lh * 8;
  const _Float16* b1r = b0r + (size_t)16 * K;

  v8f acc00 = {}, acc01 = {}, acc10 = {}, acc11 = {};
#pragma unroll 2
  for (int k0 = 0; k0 < K; k0 += 32) {
    FragU a0, a1, b0, b1;
    a0.h[0] = *(const v8h*)(a0r + k0);
    a0.h[1] = *(const v8h*)(a0r + k0 + 16);
    a1.h[0] = *(const v8h*)(a1r + k0);
    a1.h[1] = *(const v8h*)(a1r + k0 + 16);
    b0.h[0] = *(const v8h*)(b0r + k0);
    b0.h[1] = *(const v8h*)(b0r + k0 + 16);
    b1.h[0] = *(const v8h*)(b1r + k0);
    b1.h[1] = *(const v8h*)(b1r + k0 + 16);
    acc00 = WMMA_F16(a0.v, b0.v, acc00);
    acc01 = WMMA_F16(a0.v, b1.v, acc01);
    acc10 = WMMA_F16(a1.v, b0.v, acc10);
    acc11 = WMMA_F16(a1.v, b1.v, acc11);
  }

  const float dt  = ts[t] - ts[t - 1];
  const float bi0 = bias[n0 + lm];
  const float bi1 = bias[n0 + 16 + lm];
#pragma unroll
  for (int r = 0; r < 8; ++r) {
    const int m = m0 + r + 8 * lh;           // batch index (tile row 0)
    const size_t yi00 = (size_t)m * N + n0 + lm;
    const size_t yi01 = yi00 + 16;
    const size_t yi10 = (size_t)(m + 16) * N + n0 + lm;
    const size_t yi11 = yi10 + 16;
    const float v00 = y[yi00] + dt * (acc00[r] + bi0);
    const float v01 = y[yi01] + dt * (acc01[r] + bi1);
    const float v10 = y[yi10] + dt * (acc10[r] + bi0);
    const float v11 = y[yi11] + dt * (acc11[r] + bi1);
    y[yi00] = v00;  a16y[yi00] = (_Float16)v00;
    y[yi01] = v01;  a16y[yi01] = (_Float16)v01;
    y[yi10] = v10;  a16y[yi10] = (_Float16)v10;
    y[yi11] = v11;  a16y[yi11] = (_Float16)v11;
    out[((size_t)m * T_DIM + t) * X_DIM + n0 + lm]             = v00;
    out[((size_t)m * T_DIM + t) * X_DIM + n0 + 16 + lm]        = v01;
    out[((size_t)(m + 16) * T_DIM + t) * X_DIM + n0 + lm]      = v10;
    out[((size_t)(m + 16) * T_DIM + t) * X_DIM + n0 + 16 + lm] = v11;
  }
}

// ---------------------------------------------------------------------------
extern "C" void kernel_launch(void* const* d_in, const int* in_sizes, int n_in,
                              void* d_out, int out_size, void* d_ws, size_t ws_size,
                              hipStream_t stream) {
  const float* x     = (const float*)d_in[0];
  const float* ts    = (const float*)d_in[1];
  const float* W_in  = (const float*)d_in[2];
  const float* b_in  = (const float*)d_in[3];
  const float* W_h   = (const float*)d_in[4];
  const float* b_h   = (const float*)d_in[5];
  const float* W_out = (const float*)d_in[6];
  const float* b_out = (const float*)d_in[7];
  float* out = (float*)d_out;

  // Workspace layout (all offsets 256B-aligned):
  //   WT_in  : [H][X] f16      512 KB
  //   WT_h   : 3x[H][H] f16      6 MB
  //   WT_out : [X][H] f16      512 KB
  //   A16b   : [256][1024] f16 512 KB   (activation ping)
  //   A16c   : [256][1024] f16 512 KB   (activation pong)
  //   A16y   : [256][256]  f16 128 KB   (f16 state, A of first GEMM)
  //   ybuf   : [256][256]  f32 256 KB   (f32 ODE state)
  char* ws = (char*)d_ws;
  const size_t HH = (size_t)H_DIM * H_DIM;
  size_t off = 0;
  _Float16* WT_in  = (_Float16*)(ws + off); off += 524288;
  _Float16* WT_h   = (_Float16*)(ws + off); off += 6291456;
  _Float16* WT_out = (_Float16*)(ws + off); off += 524288;
  _Float16* A16b   = (_Float16*)(ws + off); off += 524288;
  _Float16* A16c   = (_Float16*)(ws + off); off += 524288;
  _Float16* A16y   = (_Float16*)(ws + off); off += 131072;
  float*    ybuf   = (float*)   (ws + off); off += 262144;

  const dim3 blk1(256);

  // One-time weight convert+transpose (stays hot in 192MB L2 afterwards)
  k_transpose_to_f16<<<(X_DIM * H_DIM + 255) / 256, blk1, 0, stream>>>(
      W_in, WT_in, X_DIM, H_DIM);
  for (int i = 0; i < NLAYERS; ++i) {
    k_transpose_to_f16<<<(H_DIM * H_DIM + 255) / 256, blk1, 0, stream>>>(
        W_h + HH * i, WT_h + HH * i, H_DIM, H_DIM);
  }
  k_transpose_to_f16<<<(H_DIM * X_DIM + 255) / 256, blk1, 0, stream>>>(
      W_out, WT_out, H_DIM, X_DIM);

  k_init<<<(B_DIM * X_DIM + 255) / 256, blk1, 0, stream>>>(x, ybuf, out, A16y);

  const dim3 blkG(128);                      // 4 waves, 2x2 wave grid
  const dim3 gridH(H_DIM / 64, B_DIM / 64);  // 16 x 4 WGs, N=1024 GEMMs
  const dim3 gridX(X_DIM / 64, B_DIM / 64);  //  4 x 4 WGs, N=256 GEMM

  for (int t = 1; t < T_DIM; ++t) {
    // h = tanh(y @ W_in + b_in)
    k_gemm_tanh<X_DIM, H_DIM><<<gridH, blkG, 0, stream>>>(A16y, WT_in, b_in, A16b);
    // three hidden layers (ping-pong A buffers)
    k_gemm_tanh<H_DIM, H_DIM><<<gridH, blkG, 0, stream>>>(A16b, WT_h,          b_h,             A16c);
    k_gemm_tanh<H_DIM, H_DIM><<<gridH, blkG, 0, stream>>>(A16c, WT_h + HH,     b_h + H_DIM,     A16b);
    k_gemm_tanh<H_DIM, H_DIM><<<gridH, blkG, 0, stream>>>(A16b, WT_h + 2 * HH, b_h + 2 * H_DIM, A16c);
    // f = h @ W_out + b_out ; y += dt*f ; out[:, t, :] = y ; A16y = f16(y)
    k_gemm_final<<<gridX, blkG, 0, stream>>>(A16c, WT_out, b_out, ts, ybuf, out, A16y, t);
  }
}
